// ExtendedAnomalyNet_7971459301527
// MI455X (gfx1250) — compile-verified, matrix operand order checked by reading, not run on hardware
//
#include <hip/hip_runtime.h>
#include <hip/hip_bf16.h>

// ---------------------------------------------------------------------------
// ExtendedAnomalyNet on MI455X (gfx1250): every layer as implicit-GEMM on
// v_wmma_f32_16x16x32_f16 (f16 in, f32 accumulate), wave32, 8 waves/WG.
// All fragment halves are 8 consecutive f16 -> b128 loads/stores; no divergent
// guards in the WMMA loops; conv1 B-tiles built cooperatively (double-
// buffered); conv2 feature map staged to LDS with
// global_load_async_to_lds_b128 (ASYNCcnt); 2x2 maxpool done with DPP
// quad_perm butterflies (pure VALU, no ds_bpermute).
// ---------------------------------------------------------------------------

typedef __attribute__((ext_vector_type(16))) _Float16 v16h;
typedef __attribute__((ext_vector_type(8)))  _Float16 v8h;
typedef __attribute__((ext_vector_type(8)))  float    v8f;

#define NP    576      // 24*24 patches
#define IMH   24
#define IMW   24

static __device__ __forceinline__ float lrelu(float v) {
  return v > 0.f ? v : v * 0.01f;
}

static __device__ __forceinline__ v8f wmma16(v16h a, v16h b, v8f c) {
  return __builtin_amdgcn_wmma_f32_16x16x32_f16(
      /*neg_a=*/false, a, /*neg_b=*/false, b,
      /*c_mod=*/(short)0, c, /*reuse_a=*/false, /*reuse_b=*/false);
}

static __device__ __forceinline__ v16h comb(v8h lo, v8h hi) {
  return __builtin_shufflevector(lo, hi, 0, 1, 2, 3, 4, 5, 6, 7,
                                         8, 9, 10, 11, 12, 13, 14, 15);
}

// max over the 4 lanes of a quad via DPP quad_perm (xor1 then xor2): the
// WMMA C-frag column index is lane&15, so columns 4q..4q+3 == one pool window.
static __device__ __forceinline__ float quad_max(float v) {
  int s  = __float_as_int(v);
  int p1 = __builtin_amdgcn_update_dpp(s, s, 0xB1, 0xF, 0xF, true);  // [1,0,3,2]
  float m = fmaxf(v, __int_as_float(p1));
  int s2 = __float_as_int(m);
  int p2 = __builtin_amdgcn_update_dpp(s2, s2, 0x4E, 0xF, 0xF, true); // [2,3,0,1]
  return fmaxf(m, __int_as_float(p2));
}

// ---------------------------------------------------------------------------
// Kernel 0: weight conversion / reordering to f16 GEMM layouts
//   W1 [128][96]   k = c*25 + di*5 + dj (pad 75->96; c==3 region is zero)
//   W2 [128][3200] k = (di*5+dj)*128 + c
//   W3 [128][3200] k = p*128 + c           (p = y*5+x, matches P2 layout)
//   W4 [256][128], W5 [128][256], WD [512][128]  (natural)
// ---------------------------------------------------------------------------
__global__ void k_prep(const float* __restrict__ c1w, const float* __restrict__ c2w,
                       const float* __restrict__ c3w, const float* __restrict__ c4w,
                       const float* __restrict__ c5w, const float* __restrict__ dw,
                       _Float16* W1, _Float16* W2, _Float16* W3,
                       _Float16* W4, _Float16* W5, _Float16* WD)
{
  const int total = 12288 + 409600 + 409600 + 32768 + 32768 + 65536;
  for (int i = blockIdx.x * blockDim.x + threadIdx.x; i < total;
       i += gridDim.x * blockDim.x) {
    int j = i;
    if (j < 12288) {                       // W1
      int m = j / 96, k = j % 96;
      float v = (k < 75) ? c1w[m * 75 + k] : 0.f;
      W1[j] = (_Float16)v;
      continue;
    }
    j -= 12288;
    if (j < 409600) {                      // W2: k = (di*5+dj)*128 + c
      int m = j / 3200, k = j % 3200;
      int c = k & 127, t = k >> 7;
      W2[j] = (_Float16)c2w[(m * 128 + c) * 25 + t];
      continue;
    }
    j -= 409600;
    if (j < 409600) {                      // W3: k = p*128 + c
      int m = j / 3200, k = j % 3200;
      int c = k & 127, p = k >> 7;
      W3[j] = (_Float16)c3w[(m * 128 + c) * 25 + p];
      continue;
    }
    j -= 409600;
    if (j < 32768)  { W4[j] = (_Float16)c4w[j]; continue; }
    j -= 32768;
    if (j < 32768)  { W5[j] = (_Float16)c5w[j]; continue; }
    j -= 32768;
    WD[j] = (_Float16)dw[j];
  }
}

// ---------------------------------------------------------------------------
// Kernel 1: conv1 (5x5, 3->128) + LeakyReLU + 2x2 maxpool, one WG per patch.
// M=128 (wave = 16-row tile), K=96 (3 wmma steps, A frags in VGPRs),
// N=784 conv positions in pool-group-of-4 order -> pooling via DPP quad max.
// The 96x16 B-tile for each N-tile is materialized ONCE by the whole WG into
// double-buffered LDS ([n][k] so frag halves are contiguous b128); the build
// gathers into registers first, then stores (no per-element DS waits).
// Patch has a 4th zero channel plane so padded k in [75,96) reads zeros
// branch-free (k/25 == 3 there).  Writes P1[patch][14][14][128] f16.
// ---------------------------------------------------------------------------
__global__ __launch_bounds__(256) void k_conv1(const float* __restrict__ x,
                                               const float* __restrict__ c1b,
                                               const _Float16* __restrict__ W1,
                                               _Float16* __restrict__ P1)
{
  __shared__ _Float16 patch[4][32][32];                 // 8 KB, plane 3 == 0
  __shared__ alignas(16) _Float16 Btile[2][16 * 96];    // 6 KB double buffer
  const int pid = blockIdx.x;
  const int oy = pid / IMW, ox = pid % IMW;
  const int tid = threadIdx.x;

  for (int i = tid; i < 4 * 32 * 32; i += 256) {
    int c = i >> 10, r = (i >> 5) & 31, col = i & 31;
    int yy = oy + r - 16, xx = ox + col - 16;    // pad = 16 each side
    float v = 0.f;
    if (c < 3 && yy >= 0 && yy < IMH && xx >= 0 && xx < IMW)
      v = x[(c * IMH + yy) * IMW + xx];
    patch[c][r][col] = (_Float16)v;
  }

  // cooperative im2col of one 16-column N-tile into Btile[buf]:
  // gather all 6 elements into registers, then store (loads pipeline).
  auto build = [&](int t, int buf) {
    _Float16 tmp[6];
#pragma unroll
    for (int q = 0; q < 6; ++q) {                // 6*256 == 16*96
      int i = q * 256 + tid;
      int n = i / 96, k = i % 96;
      int n_global = t * 16 + n;
      int p = n_global >> 2, sub = n_global & 3;
      int py = 2 * (p / 14) + (sub >> 1);
      int px = 2 * (p % 14) + (sub & 1);
      int c = k / 25, r2 = k % 25;               // c==3 -> zero plane
      tmp[q] = patch[c][py + r2 / 5][px + r2 % 5];
    }
#pragma unroll
    for (int q = 0; q < 6; ++q)
      Btile[buf][q * 256 + tid] = tmp[q];
  };

  const int wave = tid >> 5, lane = tid & 31;
  const int g = lane >> 4, ln = lane & 15;
  const int m = wave * 16 + ln;

  // Preload 3 A-fragments of the W1 row block (vectorized b128 halves)
  v16h afrag[3];
#pragma unroll
  for (int s = 0; s < 3; ++s) {
    int k0 = s * 32 + g * 8;
    afrag[s] = comb(*(const v8h*)&W1[m * 96 + k0],
                    *(const v8h*)&W1[m * 96 + k0 + 16]);
  }

  float bias[8];
#pragma unroll
  for (int r = 0; r < 8; ++r) bias[r] = c1b[wave * 16 + g * 8 + r];

  __syncthreads();           // patch ready
  build(0, 0);
  __syncthreads();           // first B-tile ready

  for (int t = 0; t < 49; ++t) {                 // 49 N-tiles = 196 pool groups
    if (t + 1 < 49) build(t + 1, (t + 1) & 1);   // overlap with consume

    const _Float16* bt = &Btile[t & 1][ln * 96];
    v8f acc = {};
#pragma unroll
    for (int s = 0; s < 3; ++s) {
      int k0 = s * 32 + g * 8;
      v16h b = comb(*(const v8h*)&bt[k0], *(const v8h*)&bt[k0 + 16]);
      acc = wmma16(afrag[s], b, acc);
    }

    // bias + LeakyReLU + 2x2 maxpool (DPP quad_perm butterflies)
    int n_global = t * 16 + ln;
    int p = n_global >> 2;
    v8h pk;
#pragma unroll
    for (int r = 0; r < 8; ++r)
      pk[r] = (_Float16)quad_max(lrelu(acc[r] + bias[r]));
    if ((lane & 3) == 0) {
      int Y = p / 14, X = p % 14;
      *(v8h*)&P1[((pid * 14 + Y) * 14 + X) * 128 + wave * 16 + g * 8] = pk;
    }
    __syncthreads();         // next tile built / this tile fully consumed
  }
}

// ---------------------------------------------------------------------------
// Kernel 2: conv2 (5x5, 128->128) + LeakyReLU + pool, one WG per patch.
// Feature map staged to LDS via global_load_async_to_lds_b128 (ASYNCcnt).
// M=128 (wave tile), K=3200 (100 k-steps, A-frag loaded once per k-step),
// N=100 positions in pool-group order (7 tiles; padded columns are clamped
// in-bounds -> harmless garbage, never stored).  All frag halves = b128.
// Writes P2[patch][p*128 + c] f16  (== conv3 im2col order).
// ---------------------------------------------------------------------------
__global__ __launch_bounds__(256) void k_conv2(const float* __restrict__ c2b,
                                               const _Float16* __restrict__ W2,
                                               const _Float16* __restrict__ P1,
                                               _Float16* __restrict__ P2)
{
  __shared__ alignas(16) _Float16 fm[14 * 14 * 128];   // 50 KB feature map
  const int pid = blockIdx.x;
  const int tid = threadIdx.x;
  const _Float16* src = P1 + pid * 14 * 14 * 128;

  // async copy global -> LDS, 16 B per lane per issue, no VGPR round-trip
  for (int i = tid * 8; i < 14 * 14 * 128; i += 256 * 8) {
    unsigned lds_off = (unsigned)(uintptr_t)&fm[i];
    const _Float16* gp = src + i;
    asm volatile("global_load_async_to_lds_b128 %0, %1, off"
                 :: "v"(lds_off), "v"(gp) : "memory");
  }
  asm volatile("s_wait_asynccnt 0" ::: "memory");
  __syncthreads();

  const int wave = tid >> 5, lane = tid & 31;
  const int g = lane >> 4, ln = lane & 15;
  const int m = wave * 16 + ln;

  // per-lane position for each of the 7 N-tiles (clamped for padded groups)
  int py[7], px[7], pool[7];
#pragma unroll
  for (int t = 0; t < 7; ++t) {
    int n_global = t * 16 + ln;
    int p = n_global >> 2, sub = n_global & 3;
    pool[t] = p;
    int pc = (p < 25) ? p : 24;                  // clamp: stay in-bounds
    py[t] = 2 * (pc / 5) + (sub >> 1);
    px[t] = 2 * (pc % 5) + (sub & 1);
  }

  v8f acc[7];
  v8f zf = {};
#pragma unroll
  for (int t = 0; t < 7; ++t) acc[t] = zf;

  const _Float16* Arow = W2 + m * 3200;
  for (int ks = 0; ks < 100; ++ks) {
    const int k0 = ks * 32 + g * 8;
    const int k1 = k0 + 16;
    __builtin_prefetch(&Arow[(ks + 1) * 32], 0, 3);   // speculative, safe
    v16h a = comb(*(const v8h*)&Arow[k0], *(const v8h*)&Arow[k1]);

    int c0 = k0 & 127, t0 = k0 >> 7, di0 = t0 / 5, dj0 = t0 % 5;
    int c1 = k1 & 127, t1 = k1 >> 7, di1 = t1 / 5, dj1 = t1 % 5;
#pragma unroll
    for (int t = 0; t < 7; ++t) {
      v8h blo = *(const v8h*)&fm[((py[t] + di0) * 14 + (px[t] + dj0)) * 128 + c0];
      v8h bhi = *(const v8h*)&fm[((py[t] + di1) * 14 + (px[t] + dj1)) * 128 + c1];
      acc[t] = wmma16(a, comb(blo, bhi), acc[t]);
    }
  }

  float bias[8];
#pragma unroll
  for (int r = 0; r < 8; ++r) bias[r] = c2b[wave * 16 + g * 8 + r];
#pragma unroll
  for (int t = 0; t < 7; ++t) {
    v8h pk;
#pragma unroll
    for (int r = 0; r < 8; ++r)
      pk[r] = (_Float16)quad_max(lrelu(acc[t][r] + bias[r]));
    if ((lane & 3) == 0 && pool[t] < 25) {
      *(v8h*)&P2[pid * 3200 + pool[t] * 128 + wave * 16 + g * 8] = pk;
    }
  }
}

// ---------------------------------------------------------------------------
// Kernel 3: conv3 (K=3200) -> conv4 (256) -> conv5 (128) -> dense (512) as
// four chained GEMMs over N = 576 patches; one WG handles 16 patches.
// Inter-layer activations kept column-major in LDS ([n][ch]) so every B-frag
// half is 8 contiguous f16 (ds_load_b128) and stores are ds_store_b128.
// Final result written straight to d_out ([512][576] == (1,512,24,24)).
// ---------------------------------------------------------------------------
__global__ __launch_bounds__(256) void k_tail(
    const float* __restrict__ c3b, const float* __restrict__ c4b,
    const float* __restrict__ c5b, const float* __restrict__ db,
    const _Float16* __restrict__ W3, const _Float16* __restrict__ W4,
    const _Float16* __restrict__ W5, const _Float16* __restrict__ WD,
    const _Float16* __restrict__ P2, float* __restrict__ out)
{
  __shared__ alignas(16) _Float16 H3[16 * 128];        // [n][ch]
  __shared__ alignas(16) _Float16 H4[16 * 256];
  __shared__ alignas(16) _Float16 H5[16 * 128];
  const int nt = blockIdx.x;                     // patch tile 0..35
  const int tid = threadIdx.x;
  const int wave = tid >> 5, lane = tid & 31;
  const int g = lane >> 4, ln = lane & 15;
  const int patch = nt * 16 + ln;

  { // ---- conv3: M=128, K=3200 ----
    int m = wave * 16 + ln;
    const _Float16* Arow = W3 + m * 3200;
    const _Float16* Brow = P2 + patch * 3200;
    v8f acc = {};
    for (int ks = 0; ks < 100; ++ks) {
      int k0 = ks * 32 + g * 8;
      v16h a = comb(*(const v8h*)&Arow[k0], *(const v8h*)&Arow[k0 + 16]);
      v16h b = comb(*(const v8h*)&Brow[k0], *(const v8h*)&Brow[k0 + 16]);
      acc = wmma16(a, b, acc);
    }
    v8h hv;
#pragma unroll
    for (int r = 0; r < 8; ++r)
      hv[r] = (_Float16)lrelu(acc[r] + c3b[wave * 16 + g * 8 + r]);
    *(v8h*)&H3[ln * 128 + wave * 16 + g * 8] = hv;
  }
  __syncthreads();

  // ---- conv4: M=256, K=128 ----
#pragma unroll
  for (int half = 0; half < 2; ++half) {
    int mt = wave + half * 8;
    int m = mt * 16 + ln;
    v8f acc = {};
#pragma unroll
    for (int ks = 0; ks < 4; ++ks) {
      int k0 = ks * 32 + g * 8;
      v16h a = comb(*(const v8h*)&W4[m * 128 + k0],
                    *(const v8h*)&W4[m * 128 + k0 + 16]);
      v16h b = comb(*(const v8h*)&H3[ln * 128 + k0],
                    *(const v8h*)&H3[ln * 128 + k0 + 16]);
      acc = wmma16(a, b, acc);
    }
    v8h hv;
#pragma unroll
    for (int r = 0; r < 8; ++r)
      hv[r] = (_Float16)lrelu(acc[r] + c4b[mt * 16 + g * 8 + r]);
    *(v8h*)&H4[ln * 256 + mt * 16 + g * 8] = hv;
  }
  __syncthreads();

  { // ---- conv5: M=128, K=256 ----
    int m = wave * 16 + ln;
    v8f acc = {};
#pragma unroll
    for (int ks = 0; ks < 8; ++ks) {
      int k0 = ks * 32 + g * 8;
      v16h a = comb(*(const v8h*)&W5[m * 256 + k0],
                    *(const v8h*)&W5[m * 256 + k0 + 16]);
      v16h b = comb(*(const v8h*)&H4[ln * 256 + k0],
                    *(const v8h*)&H4[ln * 256 + k0 + 16]);
      acc = wmma16(a, b, acc);
    }
    v8h hv;
#pragma unroll
    for (int r = 0; r < 8; ++r)
      hv[r] = (_Float16)lrelu(acc[r] + c5b[wave * 16 + g * 8 + r]);
    *(v8h*)&H5[ln * 128 + wave * 16 + g * 8] = hv;
  }
  __syncthreads();

  // ---- dense: M=512, K=128, no activation ----
#pragma unroll
  for (int q = 0; q < 4; ++q) {
    int mt = wave + q * 8;
    int m = mt * 16 + ln;
    v8f acc = {};
#pragma unroll
    for (int ks = 0; ks < 4; ++ks) {
      int k0 = ks * 32 + g * 8;
      v16h a = comb(*(const v8h*)&WD[m * 128 + k0],
                    *(const v8h*)&WD[m * 128 + k0 + 16]);
      v16h b = comb(*(const v8h*)&H5[ln * 128 + k0],
                    *(const v8h*)&H5[ln * 128 + k0 + 16]);
      acc = wmma16(a, b, acc);
    }
#pragma unroll
    for (int r = 0; r < 8; ++r) {
      int mm = mt * 16 + g * 8 + r;
      out[mm * 576 + nt * 16 + ln] = acc[r] + db[mm];
    }
  }
}

// ---------------------------------------------------------------------------
extern "C" void kernel_launch(void* const* d_in, const int* in_sizes, int n_in,
                              void* d_out, int out_size, void* d_ws, size_t ws_size,
                              hipStream_t stream) {
  (void)in_sizes; (void)n_in; (void)out_size; (void)ws_size;

  const float* x   = (const float*)d_in[0];
  const float* c1w = (const float*)d_in[1];
  const float* c1b = (const float*)d_in[2];
  const float* c2w = (const float*)d_in[3];
  const float* c2b = (const float*)d_in[4];
  const float* c3w = (const float*)d_in[5];
  const float* c3b = (const float*)d_in[6];
  const float* c4w = (const float*)d_in[7];
  const float* c4b = (const float*)d_in[8];
  const float* c5w = (const float*)d_in[9];
  const float* c5b = (const float*)d_in[10];
  const float* dw  = (const float*)d_in[11];
  const float* db  = (const float*)d_in[12];
  float* out = (float*)d_out;

  // workspace layout (bytes)
  char* ws = (char*)d_ws;
  _Float16* W1 = (_Float16*)(ws + 0);          //   24,576 B
  _Float16* W2 = (_Float16*)(ws + 24576);      //  819,200 B
  _Float16* W3 = (_Float16*)(ws + 843776);     //  819,200 B
  _Float16* W4 = (_Float16*)(ws + 1662976);    //   65,536 B
  _Float16* W5 = (_Float16*)(ws + 1728512);    //   65,536 B
  _Float16* WD = (_Float16*)(ws + 1794048);    //  131,072 B
  _Float16* P1 = (_Float16*)(ws + 1925120);    // 28,901,376 B (576*14*14*128)
  _Float16* P2 = (_Float16*)(ws + 30826496);   //  3,686,400 B (576*3200)

  k_prep<<<512, 256, 0, stream>>>(c1w, c2w, c3w, c4w, c5w, dw,
                                  W1, W2, W3, W4, W5, WD);
  k_conv1<<<NP, 256, 0, stream>>>(x, c1b, W1, P1);
  k_conv2<<<NP, 256, 0, stream>>>(c2b, W2, P1, P2);
  k_tail<<<36, 256, 0, stream>>>(c3b, c4b, c5b, db, W3, W4, W5, WD, P2, out);
}
